// MultiHeadAttention_8272107012285
// MI455X (gfx1250) — matrix operand order repaired and checked by reading, not compiled, not run on hardware
//
#include <hip/hip_runtime.h>
#include <hip/hip_bf16.h>

// ---------------------------------------------------------------------------
// MHA forward for MI455X (gfx1250, wave32, WMMA + TDM).
//   B=4, S=2048, D=1024, H=16, DK=64
// Compute-bound (~137 GFLOP vs ~200MB HBM): all matmuls through
// v_wmma_f32_16x16x32_bf16 (fp32 accum). bf16 intermediates in ws.
// K-chunk staging in the attention kernel uses the Tensor Data Mover.
// ---------------------------------------------------------------------------

typedef __bf16 bf16;
typedef __attribute__((ext_vector_type(16))) __bf16 v16bf;
typedef __attribute__((ext_vector_type(8)))  __bf16 v8bf;
typedef __attribute__((ext_vector_type(2)))  __bf16 v2bf;
typedef __attribute__((ext_vector_type(8)))  float  v8f;
typedef __attribute__((ext_vector_type(4)))  unsigned int u32x4;
typedef __attribute__((ext_vector_type(8)))  int i32x8;
typedef __attribute__((ext_vector_type(4)))  int i32x4;

#define B_   4
#define S_   2048
#define D_   1024
#define H_   16
#define DK_  64

__device__ __forceinline__ v8f wmma_bf16(v16bf a, v16bf b, v8f c) {
  return __builtin_amdgcn_wmma_f32_16x16x32_bf16(false, a, false, b, (short)0, c,
                                                 false, false);
}

// Build a v16bf fragment from two contiguous 16B (8 x bf16) runs.
__device__ __forceinline__ v16bf ld_frag2(const bf16* p0, const bf16* p1) {
  v8bf lo = *(const v8bf*)p0;
  v8bf hi = *(const v8bf*)p1;
  return __builtin_shufflevector(lo, hi, 0, 1, 2, 3, 4, 5, 6, 7,
                                 8, 9, 10, 11, 12, 13, 14, 15);
}

// A-fragment (16x32 bf16 MxK) as loaded here: lane holds M=lane&15; element j maps to
// K = 16*half + j (contiguous, same as B)  -> one contiguous run of 16.
// B-fragment (32x16 bf16 KxN): lane holds N=lane&15; element j -> K=16*half+j
//   -> one contiguous run of 16 when LDS is K-contiguous for fixed N.

// ---------------------------------------------------------------------------
// Kernel 1: QKV projections.  C[8192x1024] = A @ W + bias -> bf16 [B,H,S,DK].
// 256 threads = 8 waves; block tile 64(M) x 32(N); wave tile 16x16.
// ---------------------------------------------------------------------------
__global__ __launch_bounds__(256)
void proj_gemm(const float* __restrict__ q, const float* __restrict__ k,
               const float* __restrict__ v,
               const float* __restrict__ Wq, const float* __restrict__ bq,
               const float* __restrict__ Wk, const float* __restrict__ bk,
               const float* __restrict__ Wv, const float* __restrict__ bv,
               bf16* __restrict__ qh, bf16* __restrict__ kh,
               bf16* __restrict__ vh) {
  const float* A; const float* W; const float* bias; bf16* Out;
  if (blockIdx.z == 0)      { A = q; W = Wq; bias = bq; Out = qh; }
  else if (blockIdx.z == 1) { A = k; W = Wk; bias = bk; Out = kh; }
  else                      { A = v; W = Wv; bias = bv; Out = vh; }

  __shared__ __align__(16) bf16 Alds[64 * 32];    // row-major [m][k]
  __shared__ __align__(16) bf16 Wt[32 * 32];      // transposed [n][k]

  const int tid  = threadIdx.x;
  const int lane = tid & 31;
  const int wave = tid >> 5;
  const int wm = wave >> 1, wn = wave & 1;
  const int m0 = blockIdx.x * 64;
  const int n0 = blockIdx.y * 32;
  const int ml = lane & 15, half = lane >> 4;

  v8f acc = {};

  for (int k0 = 0; k0 < D_; k0 += 32) {
    __syncthreads();
    // Stage A tile 64x32 fp32 -> bf16, packed pair stores (4 pairs/thread).
#pragma unroll
    for (int e = 0; e < 4; ++e) {
      int elem = (tid + e * 256) * 2;
      int r = elem >> 5, c = elem & 31;
      float2 f = *(const float2*)(A + (size_t)(m0 + r) * D_ + k0 + c);
      v2bf pk; pk[0] = (bf16)f.x; pk[1] = (bf16)f.y;
      *(v2bf*)&Alds[elem] = pk;
    }
    // Stage W tile 32x32 fp32 -> bf16 TRANSPOSED: Wt[n][k] (4 elems/thread).
#pragma unroll
    for (int e = 0; e < 4; ++e) {
      int idx = tid + e * 256;
      int r = idx >> 5, c = idx & 31;     // r = k, c = n
      Wt[c * 32 + r] = (bf16)W[(size_t)(k0 + r) * D_ + n0 + c];
    }
    if (k0 + 32 < D_)
      __builtin_prefetch(&A[(size_t)(m0 + (tid >> 3)) * D_ + k0 + 32], 0, 0);
    __syncthreads();

    const bf16* arow = &Alds[(wm * 16 + ml) * 32];
    v16bf af = ld_frag2(arow + 16 * half, arow + 16 * half + 8);
    const bf16* brow = &Wt[(wn * 16 + ml) * 32 + 16 * half];
    v16bf bf_ = ld_frag2(brow, brow + 8);

    acc = wmma_bf16(af, bf_, acc);
  }

  const int col  = n0 + wn * 16 + ml;
  const float bb = bias[col];
  const int hcol = col >> 6, dk = col & 63;
#pragma unroll
  for (int r = 0; r < 8; ++r) {
    int row = m0 + wm * 16 + r + 8 * half;
    int b = row >> 11, s = row & 2047;
    Out[(size_t)(((b * H_ + hcol) * S_) + s) * DK_ + dk] = (bf16)(acc[r] + bb);
  }
}

// ---------------------------------------------------------------------------
// Kernel 2: RoPE on Q and K head buffers [B,H,S,DK], one thread per pair.
// ---------------------------------------------------------------------------
__global__ __launch_bounds__(256)
void rope_kernel(bf16* __restrict__ qh, bf16* __restrict__ kh) {
  const long idx  = (long)blockIdx.x * 256 + threadIdx.x;
  const int  p    = (int)(idx & 31);
  const long bhs  = idx >> 5;
  const int  s    = (int)(bhs & (S_ - 1));
  const float inv_freq = __expf(-((float)p / 32.0f) * 9.210340371976184f);
  float sn, cs;
  __sincosf((float)s * inv_freq, &sn, &cs);
  const long base = bhs * DK_ + 2 * p;
  float q1 = (float)qh[base], q2 = (float)qh[base + 1];
  qh[base]     = (bf16)(q1 * cs - q2 * sn);
  qh[base + 1] = (bf16)(q1 * sn + q2 * cs);
  float k1 = (float)kh[base], k2 = (float)kh[base + 1];
  kh[base]     = (bf16)(k1 * cs - k2 * sn);
  kh[base + 1] = (bf16)(k1 * sn + k2 * cs);
}

// ---------------------------------------------------------------------------
// Kernel 3: flash attention (mask all-ones -> full attention).
// Grid: (S/128, H, B); 8 waves/block; wave owns 16 q-rows x DK=64.
// K chunk staged by the Tensor Data Mover (wave 0 issues, TENSORcnt fence);
// V chunk manually transposed into LDS so PV B-fragments are vector loads.
// ---------------------------------------------------------------------------
__global__ __launch_bounds__(256)
void flash_attn(const bf16* __restrict__ qh, const bf16* __restrict__ kh,
                const bf16* __restrict__ vh, bf16* __restrict__ ao) {
  __shared__ __align__(16) bf16 Klds[32 * 64];   // [key][dk]
  __shared__ __align__(16) bf16 Vt[64 * 32];     // transposed [dk][key]
  __shared__ __align__(16) bf16 Plds[8][16 * 32];

  const int tid = threadIdx.x, lane = tid & 31, wave = tid >> 5;
  const int n = lane & 15, half = lane >> 4;
  const int b = blockIdx.z, h = blockIdx.y;
  const int q0 = blockIdx.x * 128 + wave * 16;
  const size_t bh = (size_t)(b * H_ + h) * S_ * DK_;

  // Q fragments (two 32-wide K chunks), pre-scaled by 1/sqrt(DK)=0.125.
  const bf16* Q = qh + bh + (size_t)q0 * DK_;
  v16bf qf[2];
#pragma unroll
  for (int c = 0; c < 2; ++c) {
    v16bf raw = ld_frag2(Q + n * DK_ + c * 32 + 16 * half,
                         Q + n * DK_ + c * 32 + 16 * half + 8);
#pragma unroll
    for (int j = 0; j < 16; ++j)
      qf[c][j] = (bf16)(0.125f * (float)raw[j]);
  }

  // Invariant parts of the K-chunk tensor descriptor (D#).
  const unsigned lds_k = (unsigned)(size_t)(void*)Klds;  // flat low32 = LDS off

  v8f o[4] = {{}, {}, {}, {}};
  float mrow[8], lrow[8];
#pragma unroll
  for (int r = 0; r < 8; ++r) { mrow[r] = -3.0e38f; lrow[r] = 0.f; }

  for (int kc = 0; kc < S_; kc += 32) {
    __syncthreads();
    if (wave == 0) {
      // TDM: load 32x64 bf16 K chunk (tile dim0=64 contiguous, dim1=32 rows).
      unsigned long long ga =
          (unsigned long long)(const void*)(kh + bh + (size_t)kc * DK_);
      u32x4 g0;
      g0[0] = 1u;                                      // count=1 (valid D#)
      g0[1] = lds_k;                                   // lds_addr
      g0[2] = (unsigned)ga;                            // global_addr[31:0]
      g0[3] = ((unsigned)(ga >> 32) & 0x01FFFFFFu) | (2u << 30);  // type=2
      i32x8 g1;
      g1[0] = (int)(1u << 16);          // data_size=1 (2 bytes)
      g1[1] = (int)(64u << 16);         // tensor_dim0 = 64 (low 16)
      g1[2] = (int)(2048u << 16);       // tensor_dim1 = 2048 (low 16)
      g1[3] = (int)(64u << 16);         // tile_dim0 = 64
      g1[4] = 32;                       // tile_dim1 = 32, tile_dim2 = 0
      g1[5] = 64;                       // tensor_dim0_stride = 64
      g1[6] = 0;
      g1[7] = 0;
      i32x4 z4 = {0, 0, 0, 0};
      i32x8 z8 = {0, 0, 0, 0, 0, 0, 0, 0};
      __builtin_amdgcn_tensor_load_to_lds(g0, g1, z4, z4, z8, 0);
      __builtin_amdgcn_s_wait_tensorcnt(0);
    }
    // Stage V chunk transposed: Vt[dk][key]; 8 elems per thread.
    {
      int key = tid >> 3, dk0 = (tid & 7) * 8;
      v8bf vv = *(const v8bf*)(vh + bh + (size_t)(kc + key) * DK_ + dk0);
#pragma unroll
      for (int e = 0; e < 8; ++e) Vt[(dk0 + e) * 32 + key] = vv[e];
    }
    __syncthreads();

    // Scores: two 16-key C tiles; K-dim 64 = 2 WMMAs each.
    v8f s0 = {}, s1 = {};
#pragma unroll
    for (int c = 0; c < 2; ++c) {
      const bf16* k0p = &Klds[n * 64 + c * 32 + 16 * half];
      const bf16* k1p = &Klds[(16 + n) * 64 + c * 32 + 16 * half];
      v16bf kb0 = ld_frag2(k0p, k0p + 8);
      v16bf kb1 = ld_frag2(k1p, k1p + 8);
      s0 = wmma_bf16(qf[c], kb0, s0);
      s1 = wmma_bf16(qf[c], kb1, s1);
    }

    // Online softmax (row stats uniform across each 16-lane half).
    float alpha[8];
#pragma unroll
    for (int r = 0; r < 8; ++r) {
      float mx = fmaxf(s0[r], s1[r]);
#pragma unroll
      for (int off = 1; off < 16; off <<= 1)
        mx = fmaxf(mx, __shfl_xor(mx, off, 32));
      float mnew = fmaxf(mrow[r], mx);
      alpha[r] = __expf(mrow[r] - mnew);
      float p0 = __expf(s0[r] - mnew);
      float p1 = __expf(s1[r] - mnew);
      s0[r] = p0; s1[r] = p1;
      float rs = p0 + p1;
#pragma unroll
      for (int off = 1; off < 16; off <<= 1)
        rs += __shfl_xor(rs, off, 32);
      lrow[r] = lrow[r] * alpha[r] + rs;
      mrow[r] = mnew;
    }
#pragma unroll
    for (int t = 0; t < 4; ++t)
#pragma unroll
      for (int r = 0; r < 8; ++r) o[t][r] *= alpha[r];

    // P (C layout) -> per-wave LDS -> A fragment.
    bf16* P = &Plds[wave][0];
#pragma unroll
    for (int r = 0; r < 8; ++r) {
      P[(r + 8 * half) * 32 + n]      = (bf16)s0[r];
      P[(r + 8 * half) * 32 + 16 + n] = (bf16)s1[r];
    }
    asm volatile("s_wait_dscnt 0" ::: "memory");   // in-wave LDS RAW fence
    v16bf pf = ld_frag2(&P[n * 32 + 16 * half], &P[n * 32 + 16 * half + 8]);

    // O(16x64) += P(16x32) @ V(32x64): 4 N-tiles, B-frags from Vt contiguous.
#pragma unroll
    for (int t = 0; t < 4; ++t) {
      const bf16* vp = &Vt[(t * 16 + n) * 32 + 16 * half];
      v16bf vb = ld_frag2(vp, vp + 8);
      o[t] = wmma_bf16(pf, vb, o[t]);
    }
  }

  // Normalize and write attention output in [B*S, D] layout (bf16).
#pragma unroll
  for (int r = 0; r < 8; ++r) {
    const float inv_l = 1.0f / lrow[r];
    const int row = b * S_ + q0 + r + 8 * half;
#pragma unroll
    for (int t = 0; t < 4; ++t)
      ao[(size_t)row * D_ + h * DK_ + t * 16 + n] = (bf16)(o[t][r] * inv_l);
  }
}

// ---------------------------------------------------------------------------
// Kernel 4: output projection.  out[8192x1024] = Ao(bf16) @ Wo + bo  (fp32).
// ---------------------------------------------------------------------------
__global__ __launch_bounds__(256)
void out_proj(const bf16* __restrict__ Ain, const float* __restrict__ Wo,
              const float* __restrict__ bo, float* __restrict__ out) {
  __shared__ __align__(16) bf16 Alds[64 * 32];
  __shared__ __align__(16) bf16 Wt[32 * 32];      // transposed [n][k]

  const int tid = threadIdx.x, lane = tid & 31, wave = tid >> 5;
  const int wm = wave >> 1, wn = wave & 1;
  const int m0 = blockIdx.x * 64, n0 = blockIdx.y * 32;
  const int ml = lane & 15, half = lane >> 4;

  v8f acc = {};

  for (int k0 = 0; k0 < D_; k0 += 32) {
    __syncthreads();
    {  // A tile 64x32 bf16: one 16B load/store per thread.
      int r = tid >> 2, c = (tid & 3) * 8;
      ((uint4*)Alds)[tid] =
          *(const uint4*)(Ain + (size_t)(m0 + r) * D_ + k0 + c);
    }
#pragma unroll
    for (int e = 0; e < 4; ++e) {
      int idx = tid + e * 256;
      int r = idx >> 5, c = idx & 31;
      Wt[c * 32 + r] = (bf16)Wo[(size_t)(k0 + r) * D_ + n0 + c];
    }
    __syncthreads();

    const bf16* arow = &Alds[(wm * 16 + ml) * 32];
    v16bf af = ld_frag2(arow + 16 * half, arow + 16 * half + 8);
    const bf16* brow = &Wt[(wn * 16 + ml) * 32 + 16 * half];
    v16bf bf_ = ld_frag2(brow, brow + 8);

    acc = wmma_bf16(af, bf_, acc);
  }

  const int col = n0 + wn * 16 + ml;
  const float bb = bo[col];
#pragma unroll
  for (int r = 0; r < 8; ++r) {
    int row = m0 + wm * 16 + r + 8 * half;
    out[(size_t)row * D_ + col] = acc[r] + bb;
  }
}

// ---------------------------------------------------------------------------
extern "C" void kernel_launch(void* const* d_in, const int* in_sizes, int n_in,
                              void* d_out, int out_size, void* d_ws,
                              size_t ws_size, hipStream_t stream) {
  const float* q  = (const float*)d_in[0];
  const float* k  = (const float*)d_in[1];
  const float* v  = (const float*)d_in[2];
  // d_in[3] = mask [1,1,S,S] int32: all ones in this problem -> identity.
  const float* Wq = (const float*)d_in[4];
  const float* bq = (const float*)d_in[5];
  const float* Wk = (const float*)d_in[6];
  const float* bk = (const float*)d_in[7];
  const float* Wv = (const float*)d_in[8];
  const float* bv = (const float*)d_in[9];
  const float* Wo = (const float*)d_in[10];
  const float* bo = (const float*)d_in[11];
  float* out = (float*)d_out;

  const size_t N = (size_t)B_ * S_ * D_;   // 8M elems; 4 bf16 buffers = 64MB
  bf16* qh = (bf16*)d_ws;
  bf16* kh = qh + N;
  bf16* vh = kh + N;
  bf16* ao = vh + N;

  dim3 gp((B_ * S_) / 64, D_ / 32, 3);
  proj_gemm<<<gp, 256, 0, stream>>>(q, k, v, Wq, bq, Wk, bk, Wv, bv,
                                    qh, kh, vh);

  rope_kernel<<<(B_ * H_ * S_ * 32) / 256, 256, 0, stream>>>(qh, kh);

  dim3 ga(S_ / 128, H_, B_);
  flash_attn<<<ga, 256, 0, stream>>>(qh, kh, vh, ao);

  dim3 go((B_ * S_) / 64, D_ / 32, 1);
  out_proj<<<go, 256, 0, stream>>>(ao, Wo, bo, out);
}
